// DeepRNAGenConv_72559177499330
// MI455X (gfx1250) — compile-verified, matrix operand order checked by reading, not run on hardware
//
#include <hip/hip_runtime.h>
#include <hip/hip_bf16.h>

// ---------------- problem constants ----------------
#define NB      16
#define IN_DIMS 4096
#define DD      100          // node feature dim
#define HH      200          // hidden dim
#define LL      4            // layers
#define NN      (NB * IN_DIMS)   // 65536 nodes
#define EE      (8 * NN)         // 524288 edges
#define AK      128          // K padding for GEMM1 (100 -> 128)
#define ZK      224          // K padding for GEMM2 (200 -> 224)
#define NT1     14           // col tiles GEMM1 (224/16)
#define NT2     7            // col tiles GEMM2 (112/16)
#define MT      (NN / 16)    // 4096 row tiles
#define GEPS    1e-7f

typedef __attribute__((ext_vector_type(16))) _Float16 v16h;
typedef __attribute__((ext_vector_type(8)))  _Float16 v8h;
typedef __attribute__((ext_vector_type(8)))  float    v8f;
typedef __attribute__((ext_vector_type(4)))  int      v4i;

typedef __attribute__((address_space(1))) v4i* gbl_v4i_p;
typedef __attribute__((address_space(3))) v4i* lds_v4i_p;

#if __has_builtin(__builtin_amdgcn_global_load_async_to_lds_b128) && \
    __has_builtin(__builtin_amdgcn_s_wait_asynccnt)
#define ASYNC_LDS 1
#else
#define ASYNC_LDS 0
#endif

// ---------------- small helpers ----------------
__device__ __forceinline__ float warp_sum(float v) {
    for (int off = 16; off > 0; off >>= 1) v += __shfl_xor(v, off, 32);
    return v;
}

// WMMA fragment load for 16-bit A (16xK row-major per lane) or B (packed as
// [col][K]) from a K-padded f16 buffer (global or LDS, addrspace inferred).
// CDNA5 16-bit layout: lanes 0-15 hold K {0..7,16..23}, lanes 16-31 hold
// K {8..15,24..31} -> two contiguous 16B chunks per lane.
__device__ __forceinline__ v16h load_frag16(const _Float16* base, int stride,
                                            int idx0, int k0, int lane) {
    const _Float16* p = base + (size_t)(idx0 + (lane & 15)) * stride
                             + k0 + ((lane >> 4) << 3);
    union { v16h v; v8h h[2]; } u;
    u.h[0] = *(const v8h*)(p);
    u.h[1] = *(const v8h*)(p + 16);
    return u.v;
}

// Cooperatively stage a K-padded f16 weight pack into LDS.
// Uses gfx1250 GLOBAL_LOAD_ASYNC_TO_LDS_B128 (ASYNCcnt) when available.
__device__ __forceinline__ void stage_to_lds(const _Float16* __restrict__ gsrc,
                                             _Float16* lds, int nhalfs,
                                             int tid, int nthreads) {
    int nchunks = nhalfs >> 3;   // 16B chunks
#if ASYNC_LDS
    for (int c = tid; c < nchunks; c += nthreads) {
        __builtin_amdgcn_global_load_async_to_lds_b128(
            (gbl_v4i_p)(gsrc + (size_t)c * 8),
            (lds_v4i_p)(lds + (size_t)c * 8),
            0, 0);
    }
    __builtin_amdgcn_s_wait_asynccnt(0);
#else
    for (int c = tid; c < nchunks; c += nthreads)
        *(v8h*)(lds + (size_t)c * 8) = *(const v8h*)(gsrc + (size_t)c * 8);
#endif
    __syncthreads();
}

// ---------------- weight packing ----------------
// W1pack[l][n][k], n<224, k<128 ; w1 is [L,100,200] row major
__global__ void pack_w1_kernel(const float* __restrict__ w1, _Float16* __restrict__ W1p) {
    int idx = blockIdx.x * 256 + threadIdx.x;
    if (idx >= LL * NT1 * 16 * AK) return;
    int k = idx & (AK - 1);
    int n = (idx >> 7) % (NT1 * 16);
    int l = idx / (NT1 * 16 * AK);
    float v = (k < DD && n < HH) ? w1[l * DD * HH + k * HH + n] : 0.f;
    W1p[idx] = (_Float16)v;
}

// W2pack[l][n][k], n<112, k<224 ; w2 is [L,200,100] row major
__global__ void pack_w2_kernel(const float* __restrict__ w2, _Float16* __restrict__ W2p) {
    int idx = blockIdx.x * 256 + threadIdx.x;
    if (idx >= LL * NT2 * 16 * ZK) return;
    int k = idx % ZK;
    int n = (idx / ZK) % (NT2 * 16);
    int l = idx / (NT2 * 16 * ZK);
    float v = (k < HH && n < DD) ? w2[l * HH * DD + k * DD + n] : 0.f;
    W2p[idx] = (_Float16)v;
}

// ---------------- node encoder: h = x * enc_w + enc_b ----------------
__global__ void encode_kernel(const float* __restrict__ x, const float* __restrict__ ew,
                              const float* __restrict__ eb, float* __restrict__ h) {
    int idx = blockIdx.x * 256 + threadIdx.x;
    if (idx >= NN * DD) return;
    int d = idx % DD;
    int n = idx / DD;
    h[idx] = x[n] * ew[d] + eb[d];
}

// ---------------- LayerNorm + ReLU, one wave per node ----------------
__global__ void ln_relu_kernel(const float* __restrict__ h, const float* __restrict__ g,
                               const float* __restrict__ b, float* __restrict__ hn) {
    int wave = (blockIdx.x * blockDim.x + threadIdx.x) >> 5;
    int lane = threadIdx.x & 31;
    if (wave >= NN) return;
    const float* hp = h + (size_t)wave * DD;
    float v[4]; int dcnt = 0;
    float s1 = 0.f, s2 = 0.f;
    for (int d = lane; d < DD; d += 32) {
        float t = hp[d];
        v[dcnt++] = t;
        s1 += t; s2 += t * t;
    }
    s1 = warp_sum(s1); s2 = warp_sum(s2);
    float mu  = s1 * (1.f / DD);
    float var = s2 * (1.f / DD) - mu * mu;
    float rs  = rsqrtf(var + 1e-5f);
    float* op = hn + (size_t)wave * DD;
    dcnt = 0;
    for (int d = lane; d < DD; d += 32) {
        float t = (v[dcnt++] - mu) * rs * g[d] + b[d];
        op[d] = fmaxf(t, 0.f);
    }
}

// ---------------- clear mx/den/num ----------------
__global__ void clear3_kernel(unsigned* __restrict__ mx, float* __restrict__ den,
                              float* __restrict__ num) {
    int idx = blockIdx.x * 256 + threadIdx.x;
    if (idx >= NN * DD) return;
    mx[idx] = 0u;          // all messages > 0, so uint-max over bit patterns is valid
    den[idx] = 0.f;
    num[idx] = 0.f;
}

// ---------------- scatter pass 1: segment max (wave per edge) ----------------
__global__ void scatter_max_kernel(const float* __restrict__ hn, const int* __restrict__ ei,
                                   unsigned* __restrict__ mx) {
    int e = (blockIdx.x * blockDim.x + threadIdx.x) >> 5;
    int lane = threadIdx.x & 31;
    if (e >= EE) return;
    int src = ei[e];
    int dst = ei[EE + e];
    const float* hs = hn + (size_t)src * DD;
    unsigned* md = mx + (size_t)dst * DD;
    for (int d = lane; d < DD; d += 32) {
        float m = fmaxf(hs[d], 0.f) + GEPS;
        atomicMax(md + d, __float_as_uint(m));
    }
}

// ---------------- scatter pass 2: exp-sum and weighted sum ----------------
__global__ void scatter_sum_kernel(const float* __restrict__ hn, const int* __restrict__ ei,
                                   const unsigned* __restrict__ mx,
                                   float* __restrict__ den, float* __restrict__ num) {
    int e = (blockIdx.x * blockDim.x + threadIdx.x) >> 5;
    int lane = threadIdx.x & 31;
    if (e >= EE) return;
    int src = ei[e];
    int dst = ei[EE + e];
    const float* hs = hn + (size_t)src * DD;
    const unsigned* md = mx + (size_t)dst * DD;
    float* dd = den + (size_t)dst * DD;
    float* nd = num + (size_t)dst * DD;
    for (int d = lane; d < DD; d += 32) {
        float m  = fmaxf(hs[d], 0.f) + GEPS;
        float ex = __expf(m - __uint_as_float(md[d]));
        atomicAdd(dd + d, ex);
        atomicAdd(nd + d, ex * m);
    }
}

// ---------------- finalize: A = (num/den + hn) -> f16, K-padded ----------------
__global__ void finalize_kernel(const float* __restrict__ hn, const float* __restrict__ den,
                                const float* __restrict__ num, _Float16* __restrict__ A) {
    int idx = blockIdx.x * 256 + threadIdx.x;
    if (idx >= NN * AK) return;
    int d = idx & (AK - 1);
    int n = idx >> 7;
    float a = 0.f;
    if (d < DD) {
        float dn = den[n * DD + d];
        float ag = (dn > 0.f) ? num[n * DD + d] / dn : 0.f;
        a = ag + hn[n * DD + d];
    }
    A[idx] = (_Float16)a;
}

// ---------------- GEMM1 (WMMA): Z = relu(bn(A @ W1 + b1)) -> f16 K-padded ----------------
// B (56KB) staged once per block into LDS via async copy; each wave owns one
// M-tile, keeps all A fragments in registers and sweeps all 14 N-tiles.
__global__ void __launch_bounds__(256)
gemm1_wmma_kernel(const _Float16* __restrict__ A, const _Float16* __restrict__ B,
                  const float* __restrict__ b1, const float* __restrict__ bng,
                  const float* __restrict__ bnb, _Float16* __restrict__ Z) {
    __shared__ __align__(16) _Float16 Bs[NT1 * 16 * AK];
    stage_to_lds(B, Bs, NT1 * 16 * AK, threadIdx.x, 256);

    int lane  = threadIdx.x & 31;
    int mtile = blockIdx.x * 8 + (threadIdx.x >> 5);
    int row0  = mtile * 16;
    int rbase = row0 + ((lane >> 4) << 3);

    v16h afr[AK / 32];
#pragma unroll
    for (int kk = 0; kk < AK / 32; ++kk)
        afr[kk] = load_frag16(A, AK, row0, kk * 32, lane);

    for (int ntile = 0; ntile < NT1; ++ntile) {
        int col0 = ntile * 16;
        v8f acc = {0.f, 0.f, 0.f, 0.f, 0.f, 0.f, 0.f, 0.f};
#pragma unroll
        for (int kk = 0; kk < AK / 32; ++kk) {
            v16h b = load_frag16(Bs, AK, col0, kk * 32, lane);
            acc = __builtin_amdgcn_wmma_f32_16x16x32_f16(false, afr[kk], false, b,
                                                         (short)0, acc, false, false);
        }
        int col = col0 + (lane & 15);
        float bias = 0.f, gg = 0.f, bb = 0.f;
        bool valid = (col < HH);
        if (valid) { bias = b1[col]; gg = bng[col]; bb = bnb[col]; }
#pragma unroll
        for (int v = 0; v < 8; ++v) {
            float z = valid ? fmaxf((acc[v] + bias) * gg + bb, 0.f) : 0.f;
            Z[(size_t)(rbase + v) * ZK + col] = (_Float16)z;
        }
    }
}

// ---------------- GEMM2 (WMMA): h += Z @ W2 + b2 ----------------
__global__ void __launch_bounds__(256)
gemm2_wmma_kernel(const _Float16* __restrict__ Zf, const _Float16* __restrict__ B,
                  const float* __restrict__ b2, float* __restrict__ h) {
    __shared__ __align__(16) _Float16 Bs[NT2 * 16 * ZK];
    stage_to_lds(B, Bs, NT2 * 16 * ZK, threadIdx.x, 256);

    int lane  = threadIdx.x & 31;
    int mtile = blockIdx.x * 8 + (threadIdx.x >> 5);
    int row0  = mtile * 16;
    int rbase = row0 + ((lane >> 4) << 3);

    v16h afr[ZK / 32];
#pragma unroll
    for (int kk = 0; kk < ZK / 32; ++kk)
        afr[kk] = load_frag16(Zf, ZK, row0, kk * 32, lane);

    for (int ntile = 0; ntile < NT2; ++ntile) {
        int col0 = ntile * 16;
        v8f acc = {0.f, 0.f, 0.f, 0.f, 0.f, 0.f, 0.f, 0.f};
#pragma unroll
        for (int kk = 0; kk < ZK / 32; ++kk) {
            v16h b = load_frag16(Bs, ZK, col0, kk * 32, lane);
            acc = __builtin_amdgcn_wmma_f32_16x16x32_f16(false, afr[kk], false, b,
                                                         (short)0, acc, false, false);
        }
        int col = col0 + (lane & 15);
        if (col < DD) {
            float bias = b2[col];
#pragma unroll
            for (int v = 0; v < 8; ++v) {
                size_t o = (size_t)(rbase + v) * DD + col;
                h[o] += acc[v] + bias;
            }
        }
    }
}

// ---------------- dense branch ----------------
__global__ void dense1_kernel(const float* __restrict__ x, const float* __restrict__ w,
                              const float* __restrict__ b, float* __restrict__ xs) {
    int c = blockIdx.x * 256 + threadIdx.x;
    int r = blockIdx.y;
    if (c >= 1000) return;
    const float* xr = x + (size_t)r * IN_DIMS;
    float acc = 0.f;
    for (int k = 0; k < IN_DIMS; ++k) acc = fmaf(xr[k], w[(size_t)k * 1000 + c], acc);
    acc += b[c];
    xs[r * 1000 + c] = (acc > 0.f) ? acc : 0.01f * acc;
}

__global__ void dense2_kernel(const float* __restrict__ xs, const float* __restrict__ w,
                              const float* __restrict__ b, float* __restrict__ xs2) {
    int c = threadIdx.x;
    int r = blockIdx.x;
    if (c >= DD) return;
    const float* xr = xs + (size_t)r * 1000;
    float acc = 0.f;
    for (int k = 0; k < 1000; ++k) acc = fmaf(xr[k], w[(size_t)k * DD + c], acc);
    acc += b[c];
    xs2[r * DD + c] = (acc > 0.f) ? acc : 0.01f * acc;
}

// ---------------- mean pool over contiguous graphs ----------------
__global__ void pool_kernel(const float* __restrict__ h, float* __restrict__ pooled) {
    int d = threadIdx.x;
    int b = blockIdx.x;
    if (d >= DD) return;
    float acc = 0.f;
    const float* hp = h + (size_t)b * IN_DIMS * DD + d;
    for (int i = 0; i < IN_DIMS; ++i) acc += hp[(size_t)i * DD];
    pooled[b * DD + d] = acc * (1.f / IN_DIMS);
}

// ---------------- decode: out = (0.5 xs2 + 0.5 pooled) @ dec_w + dec_b ----------------
__global__ void decode_kernel(const float* __restrict__ xs2, const float* __restrict__ pooled,
                              const float* __restrict__ dw, const float* __restrict__ db,
                              float* __restrict__ out) {
    int b = blockIdx.x;
    int lane = threadIdx.x;
    float acc = 0.f;
    for (int d = lane; d < DD; d += 32)
        acc += (0.5f * xs2[b * DD + d] + 0.5f * pooled[b * DD + d]) * dw[d];
    acc = warp_sum(acc);
    if (lane == 0) out[b] = acc + db[0];
}

// ---------------- host-side launcher ----------------
extern "C" void kernel_launch(void* const* d_in, const int* in_sizes, int n_in,
                              void* d_out, int out_size, void* d_ws, size_t ws_size,
                              hipStream_t stream) {
    const float* x     = (const float*)d_in[0];
    const int*   ei    = (const int*)  d_in[1];
    const float* enc_w = (const float*)d_in[3];
    const float* enc_b = (const float*)d_in[4];
    const float* h1_w  = (const float*)d_in[5];
    const float* h1_b  = (const float*)d_in[6];
    const float* h2_w  = (const float*)d_in[7];
    const float* h2_b  = (const float*)d_in[8];
    const float* ln_g  = (const float*)d_in[9];
    const float* ln_b  = (const float*)d_in[10];
    const float* w1    = (const float*)d_in[11];
    const float* b1    = (const float*)d_in[12];
    const float* bn_g  = (const float*)d_in[13];
    const float* bn_b  = (const float*)d_in[14];
    const float* w2    = (const float*)d_in[15];
    const float* b2    = (const float*)d_in[16];
    const float* dec_w = (const float*)d_in[17];
    const float* dec_b = (const float*)d_in[18];
    float* out = (float*)d_out;

    // workspace carve-up (256B aligned)
    char* ws = (char*)d_ws;
    size_t off = 0;
    auto carve = [&](size_t bytes) -> char* {
        char* p = ws + off;
        off += (bytes + 255) & ~(size_t)255;
        return p;
    };
    float*     h      = (float*)    carve((size_t)NN * DD * 4);
    float*     hn     = (float*)    carve((size_t)NN * DD * 4);
    unsigned*  mx     = (unsigned*) carve((size_t)NN * DD * 4);
    float*     den    = (float*)    carve((size_t)NN * DD * 4);
    float*     num    = (float*)    carve((size_t)NN * DD * 4);
    _Float16*  Af16   = (_Float16*) carve((size_t)NN * AK * 2);
    _Float16*  Zf16   = (_Float16*) carve((size_t)NN * ZK * 2);
    _Float16*  W1p    = (_Float16*) carve((size_t)LL * NT1 * 16 * AK * 2);
    _Float16*  W2p    = (_Float16*) carve((size_t)LL * NT2 * 16 * ZK * 2);
    float*     xs     = (float*)    carve((size_t)NB * 1000 * 4);
    float*     xs2    = (float*)    carve((size_t)NB * DD * 4);
    float*     pooled = (float*)    carve((size_t)NB * DD * 4);

    // weight packing (f16, WMMA B layout: [col][K])
    {
        int n1 = LL * NT1 * 16 * AK;
        pack_w1_kernel<<<(n1 + 255) / 256, 256, 0, stream>>>(w1, W1p);
        int n2 = LL * NT2 * 16 * ZK;
        pack_w2_kernel<<<(n2 + 255) / 256, 256, 0, stream>>>(w2, W2p);
    }

    // dense branch (independent of GCN chain)
    dense1_kernel<<<dim3((1000 + 255) / 256, NB), 256, 0, stream>>>(x, h1_w, h1_b, xs);
    dense2_kernel<<<NB, 128, 0, stream>>>(xs, h2_w, h2_b, xs2);

    // encoder
    encode_kernel<<<(NN * DD + 255) / 256, 256, 0, stream>>>(x, enc_w, enc_b, h);

    // GCN layers
    for (int l = 0; l < LL; ++l) {
        ln_relu_kernel<<<NN / 8, 256, 0, stream>>>(h, ln_g + l * DD, ln_b + l * DD, hn);
        clear3_kernel<<<(NN * DD + 255) / 256, 256, 0, stream>>>(mx, den, num);
        scatter_max_kernel<<<EE / 8, 256, 0, stream>>>(hn, ei, mx);
        scatter_sum_kernel<<<EE / 8, 256, 0, stream>>>(hn, ei, mx, den, num);
        finalize_kernel<<<(NN * AK + 255) / 256, 256, 0, stream>>>(hn, den, num, Af16);
        gemm1_wmma_kernel<<<MT / 8, 256, 0, stream>>>(
            Af16, W1p + (size_t)l * NT1 * 16 * AK,
            b1 + l * HH, bn_g + l * HH, bn_b + l * HH, Zf16);
        gemm2_wmma_kernel<<<MT / 8, 256, 0, stream>>>(
            Zf16, W2p + (size_t)l * NT2 * 16 * ZK, b2 + l * DD, h);
    }

    // pooling + decode
    pool_kernel<<<NB, 128, 0, stream>>>(h, pooled);
    decode_kernel<<<NB, 32, 0, stream>>>(xs2, pooled, dec_w, dec_b, out);
}
